// LlamaAttention_MUSTAFAR_75952201663292
// MI455X (gfx1250) — matrix operand order repaired and checked by reading, not compiled, not run on hardware
//
#include <hip/hip_runtime.h>
#include <hip/hip_bf16.h>
#include <float.h>

// ---------------------------------------------------------------------------
// MI455X (gfx1250) Llama attention prefill + mustafar pruning.
// GEMM-shaped work on v_wmma_f32_16x16x32_bf16 with split-bf16 (hi+lo,
// 3 WMMAs) for near-fp32 accuracy, fp32 accumulate.
// A-side operands are pre-packed once as uint32 = (bf16_hi<<16)|bf16_lo so
// the GEMM inner loop does no float->bf16 conversion on the A path.
// ---------------------------------------------------------------------------

typedef __attribute__((ext_vector_type(16))) __bf16 v16bf;
typedef __attribute__((ext_vector_type(8)))  float  v8f;

union Frag {
    v16bf v;
    unsigned short u[16];
};
static_assert(sizeof(Frag) == 32, "frag size");

__device__ __forceinline__ unsigned short bf16_bits(float f) {
    unsigned u = __float_as_uint(f);
    unsigned r = u + 0x7FFFu + ((u >> 16) & 1u);   // round-to-nearest-even
    return (unsigned short)(r >> 16);
}
__device__ __forceinline__ float bf16_val(unsigned short s) {
    return __uint_as_float(((unsigned)s) << 16);
}
__device__ __forceinline__ void split2(float x, unsigned short& hi, unsigned short& lo) {
    hi = bf16_bits(x);
    lo = bf16_bits(x - bf16_val(hi));
}
__device__ __forceinline__ unsigned packf(float x) {
    unsigned short h, l;
    split2(x, h, l);
    return (((unsigned)h) << 16) | (unsigned)l;
}

__device__ __forceinline__ v8f wmma_bf16(const Frag& a, const Frag& b, v8f c) {
    // (neg_a, A, neg_b, B, c_mod, C, reuse_a, reuse_b)
    return __builtin_amdgcn_wmma_f32_16x16x32_bf16(false, a.v, false, b.v,
                                                   (short)0, c, false, false);
}

// Problem constants
#define TT   2048
#define HIDN 4096
#define NQH  32
#define NKVH 8
#define DHD  128
#define GSZ  32
#define QKVLD 6144   // 4096 q + 1024 k + 1024 v per token

// ===========================================================================
// Pack fp32 -> uint32 (bf16_hi<<16)|bf16_lo, 4 elements/thread.
// ===========================================================================
__global__ __launch_bounds__(256)
void pack_kernel(const float* __restrict__ src, unsigned* __restrict__ dst, int n4) {
    const int i = blockIdx.x * 256 + threadIdx.x;
    if (i >= n4) return;
    const float4 x = reinterpret_cast<const float4*>(src)[i];
    uint4 o;
    o.x = packf(x.x);
    o.y = packf(x.y);
    o.z = packf(x.z);
    o.w = packf(x.w);
    reinterpret_cast<uint4*>(dst)[i] = o;
}

// ===========================================================================
// Split-bf16 WMMA GEMM:  C[M,N] = A[M,K] * W[N,K]^T
// A: pre-packed hi/lo uint32, row-major.  W: fp32 row-major (split in-loop).
// Block tile 64x128, BK=32, 128 threads = 4 waves, wave tile 32x64.
// ===========================================================================
#define BM 64
#define BN 128
#define BK 32

__global__ __launch_bounds__(128)
void gemm_splitbf16(const unsigned* __restrict__ Apk, const float* __restrict__ W,
                    float* __restrict__ C, int lda, int ldw, int ldc, int K) {
    __shared__ unsigned short As_hi[BM][BK + 8];
    __shared__ unsigned short As_lo[BM][BK + 8];
    __shared__ unsigned short Ws_hi[BN][BK + 8];
    __shared__ unsigned short Ws_lo[BN][BK + 8];

    const int tid  = threadIdx.x;
    const int wave = tid >> 5;
    const int lane = tid & 31;
    const int lh   = lane & 15;
    const int lg   = lane >> 4;
    const int bm   = blockIdx.y * BM;
    const int bn   = blockIdx.x * BN;
    const int wm   = (wave & 1) * 32;
    const int wn   = (wave >> 1) * 64;

    v8f acc[2][4] = {};

    for (int k0 = 0; k0 < K; k0 += BK) {
        __syncthreads();
        // stage A tile (64x32 packed -> hi/lo planes): 2 shifts per element
        for (int i = tid; i < (BM * BK) / 4; i += 128) {          // 4 iters
            const int r = i >> 3;
            const int c = (i & 7) << 2;
            const uint4 x = *reinterpret_cast<const uint4*>(
                &Apk[(size_t)(bm + r) * lda + k0 + c]);
            As_hi[r][c + 0] = (unsigned short)(x.x >> 16);
            As_lo[r][c + 0] = (unsigned short)(x.x);
            As_hi[r][c + 1] = (unsigned short)(x.y >> 16);
            As_lo[r][c + 1] = (unsigned short)(x.y);
            As_hi[r][c + 2] = (unsigned short)(x.z >> 16);
            As_lo[r][c + 2] = (unsigned short)(x.z);
            As_hi[r][c + 3] = (unsigned short)(x.w >> 16);
            As_lo[r][c + 3] = (unsigned short)(x.w);
        }
        // stage W tile (128x32 f32 -> hi/lo planes)
        for (int i = tid; i < (BN * BK) / 4; i += 128) {          // 8 iters
            const int r = i >> 3;
            const int c = (i & 7) << 2;
            const float4 x = *reinterpret_cast<const float4*>(
                &W[(size_t)(bn + r) * ldw + k0 + c]);
            split2(x.x, Ws_hi[r][c + 0], Ws_lo[r][c + 0]);
            split2(x.y, Ws_hi[r][c + 1], Ws_lo[r][c + 1]);
            split2(x.z, Ws_hi[r][c + 2], Ws_lo[r][c + 2]);
            split2(x.w, Ws_hi[r][c + 3], Ws_lo[r][c + 3]);
        }
        if (k0 + BK < K)
            __builtin_prefetch(&W[(size_t)(bn + tid) * ldw + k0 + BK], 0, 1);
        __syncthreads();

        // build fragments (ISA 7.12.2 layouts) and issue WMMAs
        Frag a[2][2];   // [mtile][hi/lo]
#pragma unroll
        for (int mt = 0; mt < 2; ++mt) {
            const int m = wm + mt * 16 + lh;
#pragma unroll
            for (int v = 0; v < 8; ++v) {
                const int kb = ((v < 4) ? 2 * v : 16 + 2 * (v - 4)) + lg * 8;
                a[mt][0].u[2 * v]     = As_hi[m][kb];
                a[mt][0].u[2 * v + 1] = As_hi[m][kb + 1];
                a[mt][1].u[2 * v]     = As_lo[m][kb];
                a[mt][1].u[2 * v + 1] = As_lo[m][kb + 1];
            }
        }
        Frag b[4][2];   // [ntile][hi/lo]
#pragma unroll
        for (int nt = 0; nt < 4; ++nt) {
            const int n = wn + nt * 16 + lh;
#pragma unroll
            for (int v = 0; v < 8; ++v) {
                const int kb = ((v < 4) ? 2 * v : 16 + 2 * (v - 4)) + lg * 8;
                b[nt][0].u[2 * v]     = Ws_hi[n][kb];
                b[nt][0].u[2 * v + 1] = Ws_hi[n][kb + 1];
                b[nt][1].u[2 * v]     = Ws_lo[n][kb];
                b[nt][1].u[2 * v + 1] = Ws_lo[n][kb + 1];
            }
        }
#pragma unroll
        for (int mt = 0; mt < 2; ++mt)
#pragma unroll
            for (int nt = 0; nt < 4; ++nt) {
                acc[mt][nt] = wmma_bf16(a[mt][0], b[nt][0], acc[mt][nt]);
                acc[mt][nt] = wmma_bf16(a[mt][0], b[nt][1], acc[mt][nt]);
                acc[mt][nt] = wmma_bf16(a[mt][1], b[nt][0], acc[mt][nt]);
            }
    }

    // store C (C/D layout: vgpr r -> row r + 8*lanegroup, col = lane&15)
#pragma unroll
    for (int mt = 0; mt < 2; ++mt)
#pragma unroll
        for (int nt = 0; nt < 4; ++nt)
#pragma unroll
            for (int r = 0; r < 8; ++r) {
                const int m = bm + wm + mt * 16 + r + lg * 8;
                const int n = bn + wn + nt * 16 + lh;
                C[(size_t)m * ldc + n] = acc[mt][nt][r];
            }
}

// ===========================================================================
// RoPE (in-place on qkv buffer); one thread per rotation pair.
// ===========================================================================
__global__ __launch_bounds__(256)
void rope_kernel(float* __restrict__ qkv) {
    const int idx = blockIdx.x * 256 + threadIdx.x;
    const int total = TT * (NQH + NKVH) * (DHD / 2);
    if (idx >= total) return;
    const int d2 = idx & 63;
    const int h  = (idx >> 6) % (NQH + NKVH);
    const int t  = idx / (64 * (NQH + NKVH));
    const int col = (h < NQH) ? h * DHD : HIDN + (h - NQH) * DHD;
    float* p = qkv + (size_t)t * QKVLD + col;
    // inv_freq = 10000^(-d2/64)
    const float inv = expf(-0.14391156533f * (float)d2);
    const float ang = (float)t * inv;
    const float c = cosf(ang), s = sinf(ang);
    const float x1 = p[d2], x2 = p[d2 + 64];
    p[d2]      = x1 * c - x2 * s;
    p[d2 + 64] = x2 * c + x1 * s;
}

// ===========================================================================
// Flash attention: block = (64 q-rows, 1 head); 4 waves x 16 rows.
// QK^T and P*V in split-bf16 WMMA, online softmax in fp32.
// Emits ctx in packed hi/lo form for the output projection GEMM.
// ===========================================================================
__global__ __launch_bounds__(128)
void attn_kernel(const float* __restrict__ qkv, unsigned* __restrict__ ctx_pk) {
    __shared__ unsigned short Ks_hi[32][DHD + 8];
    __shared__ unsigned short Ks_lo[32][DHD + 8];
    __shared__ unsigned short Vs_hi[32][DHD + 8];
    __shared__ unsigned short Vs_lo[32][DHD + 8];
    __shared__ unsigned short Ps_hi[4][16][40];
    __shared__ unsigned short Ps_lo[4][16][40];

    const int tid   = threadIdx.x;
    const int wave  = tid >> 5;
    const int lane  = tid & 31;
    const int lh    = lane & 15;
    const int lg    = lane >> 4;
    const int qtile = blockIdx.x;
    const int head  = blockIdx.y;
    const int kvh   = head >> 2;
    const int qbase = qtile * 64 + wave * 16;
    const float qscale = 0.08838834764831845f;   // 1/sqrt(128)

    // load Q fragments (16 rows x 128 dims, scaled) directly from global
    Frag qa[4][2];
    {
        const float* qp = qkv + (size_t)(qbase + lh) * QKVLD + head * DHD;
#pragma unroll
        for (int ks = 0; ks < 4; ++ks)
#pragma unroll
            for (int v = 0; v < 8; ++v) {
                const int kb = ((v < 4) ? 2 * v : 16 + 2 * (v - 4)) + lg * 8 + ks * 32;
                split2(qp[kb] * qscale,     qa[ks][0].u[2 * v],     qa[ks][1].u[2 * v]);
                split2(qp[kb + 1] * qscale, qa[ks][0].u[2 * v + 1], qa[ks][1].u[2 * v + 1]);
            }
    }

    v8f   o[8] = {};
    float mrun[8], lrun[8];
#pragma unroll
    for (int r = 0; r < 8; ++r) { mrun[r] = -1.0e30f; lrun[r] = 0.0f; }

    const int nkt = (qtile + 1) * 2;   // 32-key tiles covering causal range
    for (int kt = 0; kt < nkt; ++kt) {
        __syncthreads();
        // stage K/V tiles (32 keys x 128 dims) as hi/lo bf16
        for (int i = tid; i < 32 * 32; i += 128) {
            const int krow = i >> 5;
            const int c    = (i & 31) << 2;
            const size_t base = (size_t)(kt * 32 + krow) * QKVLD;
            const float4 kx = *reinterpret_cast<const float4*>(
                qkv + base + HIDN + kvh * DHD + c);
            const float4 vx = *reinterpret_cast<const float4*>(
                qkv + base + HIDN + NKVH * DHD + kvh * DHD + c);
            split2(kx.x, Ks_hi[krow][c + 0], Ks_lo[krow][c + 0]);
            split2(kx.y, Ks_hi[krow][c + 1], Ks_lo[krow][c + 1]);
            split2(kx.z, Ks_hi[krow][c + 2], Ks_lo[krow][c + 2]);
            split2(kx.w, Ks_hi[krow][c + 3], Ks_lo[krow][c + 3]);
            split2(vx.x, Vs_hi[krow][c + 0], Vs_lo[krow][c + 0]);
            split2(vx.y, Vs_hi[krow][c + 1], Vs_lo[krow][c + 1]);
            split2(vx.z, Vs_hi[krow][c + 2], Vs_lo[krow][c + 2]);
            split2(vx.w, Vs_hi[krow][c + 3], Vs_lo[krow][c + 3]);
        }
        __syncthreads();

        if (kt * 32 > qbase + 15) continue;   // wave fully above diagonal

        // --- S = Q * K^T  (16 x 32) ---
        v8f s[2] = {};
#pragma unroll
        for (int nt = 0; nt < 2; ++nt)
#pragma unroll
            for (int ks = 0; ks < 4; ++ks) {
                Frag bh, bl;
#pragma unroll
                for (int v = 0; v < 8; ++v) {
                    const int kk = ks * 32 + ((v < 4) ? 2 * v : 16 + 2 * (v - 4)) + lg * 8;
                    bh.u[2 * v]     = Ks_hi[nt * 16 + lh][kk];
                    bh.u[2 * v + 1] = Ks_hi[nt * 16 + lh][kk + 1];
                    bl.u[2 * v]     = Ks_lo[nt * 16 + lh][kk];
                    bl.u[2 * v + 1] = Ks_lo[nt * 16 + lh][kk + 1];
                }
                s[nt] = wmma_bf16(qa[ks][0], bh, s[nt]);
                s[nt] = wmma_bf16(qa[ks][0], bl, s[nt]);
                s[nt] = wmma_bf16(qa[ks][1], bh, s[nt]);
            }

        // --- causal mask + online softmax ---
        float mcur[8];
#pragma unroll
        for (int r = 0; r < 8; ++r) {
            const int grow = qbase + r + lg * 8;
            const int key0 = kt * 32 + lh;
            if (key0 > grow)      s[0][r] = -3.0e38f;
            if (key0 + 16 > grow) s[1][r] = -3.0e38f;
            mcur[r] = fmaxf(s[0][r], s[1][r]);
        }
#pragma unroll
        for (int off = 1; off < 16; off <<= 1)
#pragma unroll
            for (int r = 0; r < 8; ++r)
                mcur[r] = fmaxf(mcur[r], __shfl_xor(mcur[r], off, 32));

        float scl[8], psum[8];
#pragma unroll
        for (int r = 0; r < 8; ++r) {
            const float mn = fmaxf(mrun[r], mcur[r]);
            scl[r]  = expf(mrun[r] - mn);
            mrun[r] = mn;
            const float p0 = expf(s[0][r] - mn);
            const float p1 = expf(s[1][r] - mn);
            s[0][r] = p0; s[1][r] = p1;
            psum[r] = p0 + p1;
        }
#pragma unroll
        for (int off = 1; off < 16; off <<= 1)
#pragma unroll
            for (int r = 0; r < 8; ++r)
                psum[r] += __shfl_xor(psum[r], off, 32);
#pragma unroll
        for (int r = 0; r < 8; ++r)
            lrun[r] = lrun[r] * scl[r] + psum[r];
#pragma unroll
        for (int dt = 0; dt < 8; ++dt)
#pragma unroll
            for (int r = 0; r < 8; ++r)
                o[dt][r] *= scl[r];

        // --- stage P (D-layout -> LDS) and reload as A-fragments ---
#pragma unroll
        for (int nt = 0; nt < 2; ++nt)
#pragma unroll
            for (int r = 0; r < 8; ++r)
                split2(s[nt][r],
                       Ps_hi[wave][r + lg * 8][nt * 16 + lh],
                       Ps_lo[wave][r + lg * 8][nt * 16 + lh]);
        __builtin_amdgcn_wave_barrier();
        Frag pa, pl;
#pragma unroll
        for (int v = 0; v < 8; ++v) {
            const int kb = ((v < 4) ? 2 * v : 16 + 2 * (v - 4)) + lg * 8;
            pa.u[2 * v]     = Ps_hi[wave][lh][kb];
            pa.u[2 * v + 1] = Ps_hi[wave][lh][kb + 1];
            pl.u[2 * v]     = Ps_lo[wave][lh][kb];
            pl.u[2 * v + 1] = Ps_lo[wave][lh][kb + 1];
        }
        __builtin_amdgcn_wave_barrier();

        // --- O += P * V ---
#pragma unroll
        for (int dt = 0; dt < 8; ++dt) {
            Frag vh, vl;
#pragma unroll
            for (int v = 0; v < 8; ++v) {
                const int kk = ((v < 4) ? 2 * v : 16 + 2 * (v - 4)) + lg * 8;
                vh.u[2 * v]     = Vs_hi[kk][dt * 16 + lh];
                vh.u[2 * v + 1] = Vs_hi[kk + 1][dt * 16 + lh];
                vl.u[2 * v]     = Vs_lo[kk][dt * 16 + lh];
                vl.u[2 * v + 1] = Vs_lo[kk + 1][dt * 16 + lh];
            }
            o[dt] = wmma_bf16(pa, vh, o[dt]);
            o[dt] = wmma_bf16(pa, vl, o[dt]);
            o[dt] = wmma_bf16(pl, vh, o[dt]);
        }
    }

    // normalize, pack hi/lo, write context [T, HQ*DH]
#pragma unroll
    for (int dt = 0; dt < 8; ++dt)
#pragma unroll
        for (int r = 0; r < 8; ++r) {
            const int m = qbase + r + lg * 8;
            ctx_pk[(size_t)m * (NQH * DHD) + head * DHD + dt * 16 + lh] =
                packf(o[dt][r] / lrun[r]);
        }
}

// ===========================================================================
// Exact softmax of the last GSZ query rows per q-head, accumulated into
// rel[kvh][t] (sum over the 4 q-heads of a group). One block per kv head.
// ===========================================================================
__global__ __launch_bounds__(256)
void lastrows_kernel(const float* __restrict__ qkv, float* __restrict__ rel) {
    const int kvh = blockIdx.x;
    const int tid = threadIdx.x;
    __shared__ float qrow[DHD];
    __shared__ float red[256];
    float relacc[8];
#pragma unroll
    for (int j = 0; j < 8; ++j) relacc[j] = 0.0f;

    for (int qh = 0; qh < 4; ++qh) {
        const int head = kvh * 4 + qh;
        for (int rr = 0; rr < GSZ; ++rr) {
            const int t = TT - GSZ + rr;
            __syncthreads();
            if (tid < DHD)
                qrow[tid] = qkv[(size_t)t * QKVLD + head * DHD + tid];
            __syncthreads();

            float sv[8];
            float mx = -3.0e38f;
#pragma unroll
            for (int j = 0; j < 8; ++j) {
                const int key = tid + j * 256;
                const float* kp = qkv + (size_t)key * QKVLD + HIDN + kvh * DHD;
                float acc = 0.0f;
#pragma unroll 8
                for (int d = 0; d < DHD; ++d) acc += qrow[d] * kp[d];
                acc *= 0.08838834764831845f;
                if (key > t) acc = -3.0e38f;
                sv[j] = acc;
                mx = fmaxf(mx, acc);
            }
            red[tid] = mx; __syncthreads();
            for (int st = 128; st > 0; st >>= 1) {
                if (tid < st) red[tid] = fmaxf(red[tid], red[tid + st]);
                __syncthreads();
            }
            const float m = red[0]; __syncthreads();

            float ev[8], ssum = 0.0f;
#pragma unroll
            for (int j = 0; j < 8; ++j) { ev[j] = expf(sv[j] - m); ssum += ev[j]; }
            red[tid] = ssum; __syncthreads();
            for (int st = 128; st > 0; st >>= 1) {
                if (tid < st) red[tid] += red[tid + st];
                __syncthreads();
            }
            const float Z = red[0]; __syncthreads();
#pragma unroll
            for (int j = 0; j < 8; ++j) relacc[j] += ev[j] / Z;
        }
    }
#pragma unroll
    for (int j = 0; j < 8; ++j)
        rel[kvh * TT + tid + j * 256] = relacc[j];
}

// ===========================================================================
// Key pruning: per (kvh, t) keep |k| >= 64th-smallest |k| (stable rank 63).
// ===========================================================================
__global__ __launch_bounds__(128)
void prune_k_kernel(const float* __restrict__ qkv, float* __restrict__ out_k) {
    const int idx = blockIdx.x;
    const int kvh = idx >> 11;
    const int t   = idx & (TT - 1);
    const int d   = threadIdx.x;
    __shared__ float a[DHD];
    __shared__ float thr;

    const float kv = qkv[(size_t)t * QKVLD + HIDN + kvh * DHD + d];
    const float av = fabsf(kv);
    a[d] = av;
    __syncthreads();
    int rank = 0;
#pragma unroll 8
    for (int j = 0; j < DHD; ++j) {
        const float aj = a[j];
        rank += (aj < av) || (aj == av && j < d);
    }
    if (rank == 63) thr = av;
    __syncthreads();
    out_k[((size_t)kvh * TT + t) * DHD + d] = (av >= thr) ? kv : 0.0f;
}

// ===========================================================================
// Value pruning: score = |rel * v|; thr = ascending-sorted[64] (rank 64);
// keep score >= thr; last GSZ tokens stay dense.
// ===========================================================================
__global__ __launch_bounds__(128)
void prune_v_kernel(const float* __restrict__ qkv, const float* __restrict__ rel,
                    float* __restrict__ out_v) {
    const int idx = blockIdx.x;
    const int kvh = idx >> 11;
    const int t   = idx & (TT - 1);
    const int d   = threadIdx.x;
    __shared__ float a[DHD];
    __shared__ float thr;

    const float vv = qkv[(size_t)t * QKVLD + HIDN + NKVH * DHD + kvh * DHD + d];
    const float rl = rel[kvh * TT + t];
    const float sc = fabsf(rl * vv);
    a[d] = sc;
    __syncthreads();
    int rank = 0;
#pragma unroll 8
    for (int j = 0; j < DHD; ++j) {
        const float aj = a[j];
        rank += (aj < sc) || (aj == sc && j < d);
    }
    if (rank == 64) thr = sc;
    __syncthreads();
    float outv = (sc >= thr) ? vv : 0.0f;
    if (t >= TT - GSZ) outv = vv;   // dense residual window
    out_v[((size_t)kvh * TT + t) * DHD + d] = outv;
}

// ===========================================================================
// Launch
// ===========================================================================
extern "C" void kernel_launch(void* const* d_in, const int* in_sizes, int n_in,
                              void* d_out, int out_size, void* d_ws, size_t ws_size,
                              hipStream_t stream) {
    const float* hs = (const float*)d_in[0];
    const float* Wq = (const float*)d_in[1];
    const float* Wk = (const float*)d_in[2];
    const float* Wv = (const float*)d_in[3];
    const float* Wo = (const float*)d_in[4];
    float* out = (float*)d_out;

    float*    qkv    = (float*)d_ws;                           // [2048][6144] fp32
    unsigned* hs_pk  = (unsigned*)(qkv + (size_t)TT * QKVLD);  // [2048][4096] packed
    unsigned* ctx_pk = hs_pk + (size_t)TT * HIDN;              // [2048][4096] packed
    float*    rel    = (float*)(ctx_pk + (size_t)TT * HIDN);   // [8][2048] fp32

    float* out_attn = out;                                   // 2048*4096
    float* out_k    = out_attn + (size_t)TT * HIDN;          // 8*2048*128
    float* out_v    = out_k + (size_t)NKVH * TT * DHD;

    // Pre-pack hidden_states into uint32 (bf16_hi<<16)|bf16_lo words (done once).
    {
        const int n4 = (TT * HIDN) / 4;
        pack_kernel<<<(n4 + 255) / 256, 256, 0, stream>>>(hs, hs_pk, n4);
    }

    // QKV projections (split-bf16 WMMA GEMM, packed A)
    gemm_splitbf16<<<dim3(HIDN / BN, TT / BM), 128, 0, stream>>>(
        hs_pk, Wq, qkv, HIDN, HIDN, QKVLD, HIDN);
    gemm_splitbf16<<<dim3((NKVH * DHD) / BN, TT / BM), 128, 0, stream>>>(
        hs_pk, Wk, qkv + HIDN, HIDN, HIDN, QKVLD, HIDN);
    gemm_splitbf16<<<dim3((NKVH * DHD) / BN, TT / BM), 128, 0, stream>>>(
        hs_pk, Wv, qkv + HIDN + NKVH * DHD, HIDN, HIDN, QKVLD, HIDN);

    // RoPE on q and k
    {
        const int total = TT * (NQH + NKVH) * (DHD / 2);
        rope_kernel<<<(total + 255) / 256, 256, 0, stream>>>(qkv);
    }

    // Flash attention -> packed ctx
    attn_kernel<<<dim3(TT / 64, NQH), 128, 0, stream>>>(qkv, ctx_pk);

    // Exact last-32-row softmax -> rel
    lastrows_kernel<<<NKVH, 256, 0, stream>>>(qkv, rel);

    // Output projection (packed A = ctx)
    gemm_splitbf16<<<dim3(HIDN / BN, TT / BM), 128, 0, stream>>>(
        ctx_pk, Wo, out_attn, HIDN, HIDN, HIDN, HIDN);

    // KV pruning
    prune_k_kernel<<<NKVH * TT, DHD, 0, stream>>>(qkv, out_k);
    prune_v_kernel<<<NKVH * TT, DHD, 0, stream>>>(qkv, rel, out_v);
}
